// GRN_Encoder_9002251452441
// MI455X (gfx1250) — matrix operand order repaired and checked by reading, not compiled, not run on hardware
//
#include <hip/hip_runtime.h>
#include <math.h>
#include <stdint.h>

typedef __attribute__((ext_vector_type(16))) _Float16 v16h;
typedef __attribute__((ext_vector_type(8)))  float    v8f;

#define FDIM   128
#define T_INV  4.0f        // 1 / T_COS
#define BN_EPS 1e-5f

__device__ __forceinline__ float gelu_exact(float x) {
    return 0.5f * x * (1.0f + erff(x * 0.70710678118654752f));
}

// ---------------------------------------------------------------------------
// Weight pre-pack: B (KxN f32, row-major) -> fragment-ordered f16.
// Layout: Bp[((tn*Ksteps + ks)*32 + lane)*16 + e] =
//         (f16) B[ks*32 + (lane>=16?16:0) + e][tn*16 + (lane&15)]
// i.e. the per-lane v16h B fragment in contiguous k order (k = 16h + e), matching the A loads.
// ---------------------------------------------------------------------------
__global__ __launch_bounds__(256)
void pack_w(const float* __restrict__ B, _Float16* __restrict__ Bp, int K, int N)
{
    const int Ksteps = K >> 5;
    const int tot = (N >> 4) * Ksteps * 32;
    int idx = blockIdx.x * blockDim.x + threadIdx.x;
    if (idx >= tot) return;
    const int lane = idx & 31;
    const int ks   = (idx >> 5) % Ksteps;
    const int tn   = (idx >> 5) / Ksteps;
    const int l15   = lane & 15;
    const int bKoff = (lane >> 4) * 16;
    const float* src = B + (size_t)(ks * 32 + bKoff) * N + tn * 16 + l15;
    _Float16* dst = Bp + (size_t)idx * 16;
    #pragma unroll
    for (int e = 0; e < 16; ++e) dst[e] = (_Float16)src[(size_t)e * N];
}

// ---------------------------------------------------------------------------
// fp32 GEMM via WMMA f16 (fp32 accumulate), pre-packed f16 weights:
//   C[:, coff:coff+N] = act( A(MxK f32 row-major) @ B(KxN) + bias )
// blockDim=(32,4): 4 waves = 4 N-tiles; each wave computes 4 M-tiles (64x16),
// reusing one B fragment for 4 v_wmma per K-step. M%16==0; K%32==0; N%16==0.
// Out-of-range M-tiles: loads clamp to row M-1, stores skipped (wave-uniform),
// so EXEC is all-ones at every WMMA.
// ---------------------------------------------------------------------------
__global__ __launch_bounds__(128)
void wmma_gemm(const float* __restrict__ A, const _Float16* __restrict__ Bp,
               const float* __restrict__ bias, float* __restrict__ C,
               int M, int K, int N, int ldc, int coff, int act)
{
    const int lane = threadIdx.x;                              // 0..31
    const int tn   = blockIdx.y * blockDim.y + threadIdx.y;    // N tile
    const int tm0  = blockIdx.x * 4;                           // first of 4 M tiles
    if (tn * 16 >= N) return;                                  // wave-uniform

    const int l15    = lane & 15;
    const int hi     = lane >> 4;
    const int aKoff  = hi * 16;         // contiguous k: lane half h holds k = 16h..16h+15
    const int Ksteps = K >> 5;

    const float* arow[4];
    #pragma unroll
    for (int t = 0; t < 4; ++t) {
        int r = (tm0 + t) * 16 + l15;
        if (r >= M) r = M - 1;                 // clamp: loads stay in-bounds
        arow[t] = A + (size_t)r * K;
    }

    const v16h* bp = (const v16h*)Bp + (size_t)tn * Ksteps * 32 + lane;

    v8f acc[4] = {};
    for (int ks = 0; ks < Ksteps; ++ks) {
        const v16h b = bp[(size_t)ks * 32];    // 32B contiguous (2x b128)
        const int k0 = ks * 32;
        __builtin_prefetch(arow[0] + k0 + 32, 0, 1);
        #pragma unroll
        for (int t = 0; t < 4; ++t) {
            const float* ar = arow[t] + k0;
            const float4 a0 = *(const float4*)(ar + aKoff);
            const float4 a1 = *(const float4*)(ar + aKoff + 4);
            const float4 a2 = *(const float4*)(ar + aKoff + 8);
            const float4 a3 = *(const float4*)(ar + aKoff + 12);
            v16h a;
            a[0]=(_Float16)a0.x; a[1]=(_Float16)a0.y; a[2]=(_Float16)a0.z; a[3]=(_Float16)a0.w;
            a[4]=(_Float16)a1.x; a[5]=(_Float16)a1.y; a[6]=(_Float16)a1.z; a[7]=(_Float16)a1.w;
            a[8]=(_Float16)a2.x; a[9]=(_Float16)a2.y; a[10]=(_Float16)a2.z; a[11]=(_Float16)a2.w;
            a[12]=(_Float16)a3.x; a[13]=(_Float16)a3.y; a[14]=(_Float16)a3.z; a[15]=(_Float16)a3.w;
            acc[t] = __builtin_amdgcn_wmma_f32_16x16x32_f16(false, a, false, b,
                                                            (short)0, acc[t], false, false);
        }
    }

    const float bv = bias ? bias[tn * 16 + l15] : 0.0f;
    #pragma unroll
    for (int t = 0; t < 4; ++t) {
        if ((tm0 + t) * 16 >= M) break;        // wave-uniform
        float* crow = C + (size_t)((tm0 + t) * 16 + hi * 8) * ldc + coff + tn * 16 + l15;
        #pragma unroll
        for (int r = 0; r < 8; ++r) {
            float v = acc[t][r] + bv;
            if (act) v = gelu_exact(v);
            crow[(size_t)r * ldc] = v;
        }
    }
}

// ---------------------------------------------------------------------------
// Elementwise / reduction helpers
// ---------------------------------------------------------------------------
__global__ void xaux_kernel(const float* __restrict__ aux, const float* __restrict__ c,
                            const float* __restrict__ d, float* __restrict__ xaux, int n)
{
    int i = blockIdx.x * blockDim.x + threadIdx.x;
    if (i < n) {
        float t = c[0] * aux[i] - d[0];
        xaux[i] = 1.0f / (1.0f + expf(-t));
    }
}

// Coalesced BN partial sums: 256 threads = 2 rows x 128 cols per step.
__global__ __launch_bounds__(256)
void bn_partial(const float* __restrict__ h, float* __restrict__ s1,
                float* __restrict__ s2, int n)
{
    const int col   = threadIdx.x & 127;
    const int rhalf = threadIdx.x >> 7;
    float a = 0.f, b = 0.f;
    for (int r = blockIdx.x * 2 + rhalf; r < n; r += gridDim.x * 2) {
        float v = h[(size_t)r * FDIM + col];
        a += v; b += v * v;
    }
    __shared__ float sa[256], sb[256];
    sa[threadIdx.x] = a; sb[threadIdx.x] = b;
    __syncthreads();
    if (threadIdx.x < 128) {
        a = sa[threadIdx.x] + sa[threadIdx.x + 128];
        b = sb[threadIdx.x] + sb[threadIdx.x + 128];
        atomicAdd(s1 + threadIdx.x, a);
        atomicAdd(s2 + threadIdx.x, b);
    }
}

__global__ void bn_finalize(const float* __restrict__ s1, const float* __restrict__ s2,
                            float* __restrict__ mu, float* __restrict__ var, int n)
{
    int c = threadIdx.x;
    if (c < FDIM) {
        float m = s1[c] / (float)n;
        mu[c]  = m;
        var[c] = s2[c] / (float)n - m * m;
    }
}

__global__ void bn_apply(const float* __restrict__ h, const float* __restrict__ mu,
                         const float* __restrict__ var, const float* __restrict__ gamma,
                         const float* __restrict__ beta, float* __restrict__ out, int n)
{
    size_t idx = (size_t)blockIdx.x * blockDim.x + threadIdx.x;
    if (idx < (size_t)n * FDIM) {
        int c = (int)(idx & (FDIM - 1));
        float v = (h[idx] - mu[c]) * rsqrtf(var[c] + BN_EPS);
        out[idx] = v * gamma[c] + beta[c];
    }
}

__global__ __launch_bounds__(256)
void row_rinv(const float* __restrict__ X, float* __restrict__ rinv, int n)
{
    int row = blockIdx.x * blockDim.y + threadIdx.y;
    if (row >= n) return;
    const float4 v = ((const float4*)(X + (size_t)row * FDIM))[threadIdx.x];
    float s = v.x * v.x + v.y * v.y + v.z * v.z + v.w * v.w;
    for (int o = 16; o > 0; o >>= 1) s += __shfl_down(s, o, 32);
    if (threadIdx.x == 0) rinv[row] = 1.0f / fmaxf(sqrtf(s), 1e-12f);
}

__global__ void add_col_bias(float* __restrict__ X, const float* __restrict__ b, int n)
{
    size_t idx = (size_t)blockIdx.x * blockDim.x + threadIdx.x;
    if (idx < (size_t)n * FDIM) X[idx] += b[idx & (FDIM - 1)];
}

// ---------------------------------------------------------------------------
// Edge passes (segment softmax over incoming edges of node i, then scatter).
// ei[0..E-1]=src, ei[E..2E-1]=dst; edges E..E2-1 are self loops.
// swap==0: j=src,i=dst ; swap==1: j=dst,i=src.
// ---------------------------------------------------------------------------
__global__ __launch_bounds__(256)
void edge_pass1(const int* __restrict__ ei, int E, int E2, int swap,
                const float* __restrict__ xl, const float* __restrict__ xr,
                const float* __restrict__ rl, const float* __restrict__ rr,
                const float* __restrict__ xaux,
                float* __restrict__ araw, float* __restrict__ segm)
{
    int e = blockIdx.x * blockDim.y + threadIdx.y;
    if (e >= E2) return;
    int s0, d0;
    if (e < E) { s0 = ei[e]; d0 = ei[e + E]; } else { s0 = d0 = e - E; }
    const int j = swap ? d0 : s0;
    const int i = swap ? s0 : d0;
    const float4 u = ((const float4*)(xr + (size_t)i * FDIM))[threadIdx.x];
    const float4 v = ((const float4*)(xl + (size_t)j * FDIM))[threadIdx.x];
    float dot = u.x * v.x + u.y * v.y + u.z * v.z + u.w * v.w;
    for (int o = 16; o > 0; o >>= 1) dot += __shfl_down(dot, o, 32);
    if (threadIdx.x == 0) {
        float score = fabsf(dot) * rl[j] * rr[i];        // |<n_i, n_j>|
        float raw   = xaux[j] * score * T_INV;           // >= 0 always
        araw[e] = raw;
        atomicMax((unsigned int*)(segm + i), __float_as_uint(raw));
    }
}

__global__ void edge_pass2(const int* __restrict__ ei, int E, int E2, int swap,
                           float* __restrict__ araw, const float* __restrict__ segm,
                           float* __restrict__ segs)
{
    int e = blockIdx.x * blockDim.x + threadIdx.x;
    if (e >= E2) return;
    int i;
    if (e < E) i = swap ? ei[e] : ei[e + E];
    else       i = e - E;
    float ex = expf(araw[e] - segm[i]);
    araw[e] = ex;
    atomicAdd(segs + i, ex);
}

__global__ __launch_bounds__(256)
void edge_pass3(const int* __restrict__ ei, int E, int E2, int swap,
                const float* __restrict__ xl, const float* __restrict__ araw,
                const float* __restrict__ segs, float* __restrict__ agg)
{
    int e = blockIdx.x * blockDim.y + threadIdx.y;
    if (e >= E2) return;
    int s0, d0;
    if (e < E) { s0 = ei[e]; d0 = ei[e + E]; } else { s0 = d0 = e - E; }
    const int j = swap ? d0 : s0;
    const int i = swap ? s0 : d0;
    const float w = araw[e] / segs[i];
    const float4 v = ((const float4*)(xl + (size_t)j * FDIM))[threadIdx.x];
    float* o = agg + (size_t)i * FDIM + threadIdx.x * 4;
    atomicAdd(o + 0, v.x * w);
    atomicAdd(o + 1, v.y * w);
    atomicAdd(o + 2, v.z * w);
    atomicAdd(o + 3, v.w * w);
}

// ---------------------------------------------------------------------------
extern "C" void kernel_launch(void* const* d_in, const int* in_sizes, int n_in,
                              void* d_out, int out_size, void* d_ws, size_t ws_size,
                              hipStream_t stream)
{
    (void)n_in; (void)out_size; (void)ws_size;
    const float* x     = (const float*)d_in[0];
    const int*   ei    = (const int*)  d_in[1];
    const float* aux   = (const float*)d_in[2];
    const float* cpar  = (const float*)d_in[3];
    const float* dpar  = (const float*)d_in[4];
    const float* w_in  = (const float*)d_in[5];
    const float* b_in  = (const float*)d_in[6];
    const float* bn_g  = (const float*)d_in[7];
    const float* bn_b  = (const float*)d_in[8];
    const float* lin_l = (const float*)d_in[9];
    const float* lin_r = (const float*)d_in[10];
    const float* attb  = (const float*)d_in[11];
    const float* wcat  = (const float*)d_in[12];
    const float* fw1   = (const float*)d_in[13];
    const float* fb1   = (const float*)d_in[14];
    const float* fw2   = (const float*)d_in[15];
    const float* fb2   = (const float*)d_in[16];
    const float* wproj = (const float*)d_in[17];
    const float* bproj = (const float*)d_in[18];

    const int N    = in_sizes[2];             // 50000 nodes
    const int IND  = in_sizes[0] / N;         // 256
    const int E    = in_sizes[1] / 2;         // 800000
    const int E2   = E + N;                   // + self loops
    const int OUTN = in_sizes[18];            // 512

    // ---- fp32 workspace carve-out (every chunk a multiple of 4 floats)
    float* p = (float*)d_ws;
    auto alloc = [&](size_t elems) { float* r = p; p += elems; return r; };
    float* xaux  = alloc(N);
    float* h     = alloc((size_t)N * FDIM);
    float* hbn   = alloc((size_t)N * FDIM);
    float* xl[2] = { alloc((size_t)N * FDIM), alloc((size_t)N * FDIM) };
    float* xr[2] = { alloc((size_t)N * FDIM), alloc((size_t)N * FDIM) }; // reused as agg
    float* rl[2] = { alloc(N), alloc(N) };
    float* rr[2] = { alloc(N), alloc(N) };
    float* segm  = alloc(N);
    float* segs  = alloc(N);
    float* araw  = alloc(E2);
    float* z     = alloc((size_t)N * 2 * FDIM);
    float* t     = alloc((size_t)N * FDIM);
    float* mu    = alloc(FDIM);
    float* var   = alloc(FDIM);
    float* bs1   = alloc(FDIM);
    float* bs2   = alloc(FDIM);

    // ---- fp16 packed-weight pool (align to 256B)
    _Float16* hp = (_Float16*)(((uintptr_t)p + 255) & ~(uintptr_t)255);
    auto allocH = [&](size_t elems) { _Float16* r = hp; hp += elems; return r; };
    _Float16* pw_in  = allocH((size_t)IND * FDIM);
    _Float16* plinl[4]; _Float16* plinr[4]; _Float16* pwcat[4];
    for (int i = 0; i < 4; ++i) { plinl[i] = allocH(FDIM * FDIM);
                                  plinr[i] = allocH(FDIM * FDIM);
                                  pwcat[i] = allocH(FDIM * FDIM); }
    _Float16* pfw1[2]; _Float16* pfw2[2];
    for (int i = 0; i < 2; ++i) { pfw1[i] = allocH(2 * FDIM * FDIM);
                                  pfw2[i] = allocH(FDIM * FDIM); }
    _Float16* pwproj = allocH((size_t)FDIM * OUTN);

    auto pack = [&](const float* B, _Float16* Bp, int K, int Nc) {
        int tot = (Nc / 16) * (K / 32) * 32;
        hipLaunchKernelGGL(pack_w, dim3((tot + 255) / 256), dim3(256), 0, stream, B, Bp, K, Nc);
    };
    auto gemm = [&](const float* A, const _Float16* Bp, const float* bias, float* C,
                    int M, int K, int Nc, int ldc, int coff, int act) {
        dim3 blk(32, 4);
        int mt = (M + 15) / 16;
        dim3 grd((unsigned)((mt + 3) / 4), (unsigned)((Nc / 16 + 3) / 4));
        hipLaunchKernelGGL(wmma_gemm, grd, blk, 0, stream,
                           A, Bp, bias, C, M, K, Nc, ldc, coff, act);
    };

    // ---- pack all weights (deterministic, every call)
    pack(w_in, pw_in, IND, FDIM);
    for (int i = 0; i < 4; ++i) {
        pack(lin_l + (size_t)i * FDIM * FDIM, plinl[i], FDIM, FDIM);
        pack(lin_r + (size_t)i * FDIM * FDIM, plinr[i], FDIM, FDIM);
        pack(wcat  + (size_t)i * FDIM * FDIM, pwcat[i], FDIM, FDIM);
    }
    for (int i = 0; i < 2; ++i) {
        pack(fw1 + (size_t)i * 2 * FDIM * FDIM, pfw1[i], 2 * FDIM, FDIM);
        pack(fw2 + (size_t)i * FDIM * FDIM,     pfw2[i], FDIM,     FDIM);
    }
    pack(wproj, pwproj, FDIM, OUTN);

    // x_aux = sigmoid(c*aux - d)
    hipLaunchKernelGGL(xaux_kernel, dim3((N + 255) / 256), dim3(256), 0, stream,
                       aux, cpar, dpar, xaux, N);
    // h = x @ w_in + b_in
    gemm(x, pw_in, b_in, h, N, IND, FDIM, FDIM, 0, 0);

    const size_t ew = (size_t)N * FDIM;
    const dim3 eblk(32, 8);
    const dim3 egrd((E2 + 7) / 8);

    for (int l = 0; l < 2; ++l) {
        hipMemsetAsync(bs1, 0, FDIM * sizeof(float), stream);
        hipMemsetAsync(bs2, 0, FDIM * sizeof(float), stream);
        hipLaunchKernelGGL(bn_partial, dim3(256), dim3(256), 0, stream, h, bs1, bs2, N);
        hipLaunchKernelGGL(bn_finalize, dim3(1), dim3(FDIM), 0, stream, bs1, bs2, mu, var, N);
        hipLaunchKernelGGL(bn_apply, dim3((unsigned)((ew + 255) / 256)), dim3(256), 0, stream,
                           h, mu, var, bn_g + l * FDIM, bn_b + l * FDIM, hbn, N);

        for (int dir = 0; dir < 2; ++dir) {
            const int wi = l * 2 + dir;
            gemm(hbn, plinl[wi], nullptr, xl[dir], N, FDIM, FDIM, FDIM, 0, 0);
            gemm(hbn, plinr[wi], nullptr, xr[dir], N, FDIM, FDIM, FDIM, 0, 0);
            hipLaunchKernelGGL(row_rinv, dim3((N + 7) / 8), eblk, 0, stream, xl[dir], rl[dir], N);
            hipLaunchKernelGGL(row_rinv, dim3((N + 7) / 8), eblk, 0, stream, xr[dir], rr[dir], N);

            hipMemsetAsync(segm, 0, N * sizeof(float), stream);
            hipMemsetAsync(segs, 0, N * sizeof(float), stream);
            hipLaunchKernelGGL(edge_pass1, egrd, eblk, 0, stream,
                               ei, E, E2, dir, xl[dir], xr[dir], rl[dir], rr[dir],
                               xaux, araw, segm);

            float* agg = xr[dir];                       // xr no longer needed
            hipMemsetAsync(agg, 0, ew * sizeof(float), stream);
            hipLaunchKernelGGL(edge_pass2, dim3((E2 + 255) / 256), dim3(256), 0, stream,
                               ei, E, E2, dir, araw, segm, segs);
            hipLaunchKernelGGL(edge_pass3, egrd, eblk, 0, stream,
                               ei, E, E2, dir, xl[dir], araw, segs, agg);

            hipLaunchKernelGGL(add_col_bias, dim3((unsigned)((ew + 255) / 256)), dim3(256),
                               0, stream, agg, attb + (size_t)wi * FDIM, N);
            // z[:, dir*128 : dir*128+128] = gelu(agg @ wcat)
            gemm(agg, pwcat[wi], nullptr, z, N, FDIM, FDIM, 2 * FDIM, dir * FDIM, 1);
        }
        // t = gelu(z @ ffn_w1 + b1) ; h = t @ ffn_w2 + b2
        gemm(z, pfw1[l], fb1 + l * FDIM, t, N, 2 * FDIM, FDIM, FDIM, 0, 1);
        gemm(t, pfw2[l], fb2 + l * FDIM, h, N, FDIM,     FDIM, FDIM, 0, 0);
    }
    // out = h @ w_proj + b_proj
    gemm(h, pwproj, bproj, (float*)d_out, N, FDIM, OUTN, OUTN, 0, 0);
}
